// SNN_42219528519844
// MI455X (gfx1250) — matrix operand order repaired and checked by reading, not compiled, not run on hardware
//
#include <hip/hip_runtime.h>

// ---------------------------------------------------------------------------
// SNN forward, MI455X (gfx1250): fused per-layer [GEMM over T] + membrane scan.
// A = spikes (exact in f16), B = W rounded to f16, f32 accumulate via
// v_wmma_f32_16x16x32_f16 (wave32). A tiles are DMA'd into LDS by the Tensor
// Data Mover (tensor_load_to_lds, TENSORcnt), double-buffered against WMMA.
// ---------------------------------------------------------------------------

typedef __attribute__((ext_vector_type(16))) _Float16 v16h;
typedef __attribute__((ext_vector_type(8)))  float    v8f;
typedef __attribute__((ext_vector_type(4)))  unsigned int v4u;
typedef __attribute__((ext_vector_type(8)))  int      v8i;
typedef __attribute__((ext_vector_type(4)))  int      v4i;

union F16x16 { v16h v; uint4 q[2]; };

#define SNN_T 50
#define SNN_B 256

// f32 -> f16 conversion (prepass for spikes + weights)
__global__ void snn_cvt_f16(const float* __restrict__ x, _Float16* __restrict__ y, size_t n) {
  size_t i = (size_t)blockIdx.x * blockDim.x + threadIdx.x;
  size_t stride = (size_t)gridDim.x * blockDim.x;
  for (; i < n; i += stride) y[i] = (_Float16)x[i];
}

// TDM: DMA a [64 rows x 64 halves] f16 tile (row stride IN halves in memory)
// into LDS at byte offset lds_off, inserting 16B padding after every 128B row
// (=> LDS row stride 72 halves). data_size=2B. type=2 ("image").
__device__ __forceinline__ void tdm_load_tile_a(const _Float16* gptr, unsigned lds_off, int in_halves) {
  unsigned long long ga = (unsigned long long)(uintptr_t)gptr;
  v4u g0;
  g0.x = 1u;                                       // count=1 (valid), gather off
  g0.y = lds_off;                                  // lds_addr
  g0.z = (unsigned)ga;                             // global_addr[31:0]
  g0.w = (unsigned)((ga >> 32) & 0x1FFFFFFull)     // global_addr[56:32]
         | (2u << 30);                             // type=2
  v8i g1;
  g1[0] = (int)((1u << 16)                         // data_size = 2 bytes
              | (1u << 20)                         // pad_enable
              | (4u << 22)                         // pad_interval: 16x8B = 128B
              | (3u << 25));                       // pad_amount: 4 DWORDs = 16B
  g1[1] = (int)(((unsigned)in_halves & 0xFFFFu) << 16);          // tensor_dim0[15:0]
  g1[2] = (int)((((unsigned)in_halves >> 16) & 0xFFFFu)          // tensor_dim0[31:16]
              | (256u << 16));                                   // tensor_dim1[15:0] = B
  g1[3] = (int)(64u << 16);                        // tensor_dim1 hi=0 | tile_dim0=64
  g1[4] = 64;                                      // tile_dim1=64 rows | tile_dim2=0
  g1[5] = in_halves;                               // tensor_dim0_stride[31:0]
  g1[6] = 0;                                       // stride hi | dim1_stride lo
  g1[7] = 0;
  v4i z4 = {};
  v8i z8 = {};
  __builtin_amdgcn_tensor_load_to_lds(g0, g1, z4, z4, z8, 0);
}

// One SNN layer, fused GEMM + scan.
//   X   : [T, B, IN]  f16 spikes (0/1)
//   Wh  : [OUT, IN]   f16 weights (row-major)
//   bias: [OUT]       f32
//   memI: [B, OUT]    f32 initial membrane
// MODE 0: write spikes f16 to Yh [T,B,OUT]
// MODE 1: write spikes f32 to Yf [T,B,OUT] and h = mean_t to Hout [B,OUT]
//
// Block: 128 threads = 4 waves (2 along M x 2 along N), WG tile 64(b) x 64(out).
// LDS: W slice [64][IN+8] f16 (reused across all t) + 2x A tile [64][72] f16.
template <int MODE, int IN, int OUT>
__global__ __launch_bounds__(128)
void snn_layer_kernel(const _Float16* __restrict__ X,
                      const _Float16* __restrict__ Wh,
                      const float*    __restrict__ bias,
                      const float*    __restrict__ memI,
                      _Float16*       __restrict__ Yh,
                      float*          __restrict__ Yf,
                      float*          __restrict__ Hout)
{
  constexpr int WROW = IN + 8;           // padded W row in halves (16B align kept)
  constexpr int KCH  = IN >> 6;          // number of 64-wide K chunks
  constexpr int JTOT = SNN_T * KCH;

  extern __shared__ _Float16 lds[];
  _Float16* Wl  = lds;                   // 64 * WROW halves
  _Float16* Al0 = lds + 64 * WROW;       // 64 * 72 halves (buffer 0)
  _Float16* Al1 = Al0 + 64 * 72;         // 64 * 72 halves (buffer 1)

  const unsigned ldsbase = __builtin_amdgcn_groupstaticsize();  // dynamic-LDS base
  const unsigned aoff0 = ldsbase + (unsigned)(64 * WROW) * 2u;
  const unsigned aoff1 = aoff0 + (unsigned)(64 * 72) * 2u;

  const int tid  = threadIdx.x;
  const int lane = tid & 31;
  const int wid  = tid >> 5;
  const int wm   = wid & 1;              // wave position along M
  const int wn   = wid >> 1;             // wave position along N
  const int ln15 = lane & 15;
  const int lhi  = lane >> 4;            // 0 or 1

  const int NB0 = blockIdx.x * 64;       // out-column block
  const int MB0 = blockIdx.y * 64;       // batch-row block

  const _Float16* Xblk = X + (size_t)MB0 * IN;   // row MB0 of every timestep slab

  // ---- prologue: kick off DMA of A tile j=0 while we set up ----------------
  if (wid == 0) tdm_load_tile_a(Xblk, aoff0, IN);

  // ---- stage W slice [64 x IN] into LDS (coalesced b128, one-time) ---------
  {
    constexpr int cpr = IN >> 3;         // 16B chunks per row
    for (int g = tid; g < 64 * cpr; g += 128) {
      int row = g / cpr;
      int ch  = g - row * cpr;
      const uint4* src = (const uint4*)(Wh + (size_t)(NB0 + row) * IN) + ch;
      *((uint4*)(Wl + row * WROW + ch * 8)) = *src;
    }
  }

  // ---- per-lane bias + membrane state (registers, persist over t) ----------
  float bv[2];
  float m [2][2][8];
  float hs[2][2][8];
  for (int ni = 0; ni < 2; ++ni)
    bv[ni] = bias[NB0 + wn * 32 + ni * 16 + ln15];
  for (int mi = 0; mi < 2; ++mi)
    for (int ni = 0; ni < 2; ++ni) {
      const int N = NB0 + wn * 32 + ni * 16 + ln15;
      for (int v = 0; v < 8; ++v) {
        const int Mr = MB0 + wm * 32 + mi * 16 + v + lhi * 8;
        m[mi][ni][v] = memI[(size_t)Mr * OUT + N];
        if (MODE) hs[mi][ni][v] = 0.0f;
      }
    }

  v8f acc[2][2];
  {
    v8f z = {};
    for (int mi = 0; mi < 2; ++mi)
      for (int ni = 0; ni < 2; ++ni)
        acc[mi][ni] = z;
  }

  for (int j = 0; j < JTOT; ++j) {
    const int t  = j / KCH;              // KCH is a power of two -> shifts
    const int kc = j - t * KCH;

    // All waves finished reading buffer (j-1)&1 == (j+1)&1 -> safe to refill.
    __syncthreads();
    if (wid == 0) {
      if (j + 1 < JTOT) {
        const int t1  = (j + 1) / KCH;
        const int kc1 = (j + 1) - t1 * KCH;
        tdm_load_tile_a(Xblk + (size_t)t1 * SNN_B * IN + kc1 * 64,
                        ((j + 1) & 1) ? aoff1 : aoff0, IN);
        __builtin_amdgcn_s_wait_tensorcnt(1);   // tile j complete (in-order TDM)
      } else {
        __builtin_amdgcn_s_wait_tensorcnt(0);
      }
    }
    __syncthreads();                     // tile j visible to all waves

    const _Float16* Ab = (j & 1) ? Al1 : Al0;

    for (int ks = 0; ks < 2; ++ks) {
      // A fragments: lane holds row M, halves K+{0..7} and K+{16..23}; lhi adds +8
      F16x16 a[2], b[2];
      for (int mi = 0; mi < 2; ++mi) {
        const _Float16* p = Ab + (wm * 32 + mi * 16 + ln15) * 72 + ks * 32 + lhi * 8;
        a[mi].q[0] = *(const uint4*)(p);
        a[mi].q[1] = *(const uint4*)(p + 16);
      }
      // B fragments: lane holds col N (= W row), halves K+{0..7} and K+{16..23}; lhi adds +8
      for (int ni = 0; ni < 2; ++ni) {
        const _Float16* p = Wl + (wn * 32 + ni * 16 + ln15) * WROW + kc * 64 + ks * 32 + lhi * 8;
        b[ni].q[0] = *(const uint4*)(p);
        b[ni].q[1] = *(const uint4*)(p + 16);
      }
      for (int mi = 0; mi < 2; ++mi)
        for (int ni = 0; ni < 2; ++ni)
          acc[mi][ni] = __builtin_amdgcn_wmma_f32_16x16x32_f16(
              false, a[mi].v, false, b[ni].v,
              (short)0, acc[mi][ni], false, false);
    }

    if (kc == KCH - 1) {
      // ---- membrane scan + spike emit (sum_u path is grad-only -> dead) ----
      for (int mi = 0; mi < 2; ++mi)
        for (int ni = 0; ni < 2; ++ni) {
          const int N = NB0 + wn * 32 + ni * 16 + ln15;
          for (int v = 0; v < 8; ++v) {
            const float du = acc[mi][ni][v] + bv[ni];
            float mm = m[mi][ni][v] + du;
            const float s = (mm > 15.0f) ? 1.0f : 0.0f;
            mm = (mm > 15.0f) ? 0.0f : fmaxf(mm, 0.0f); // reset on spike, clip low
            m[mi][ni][v] = mm;
            const int Mr = MB0 + wm * 32 + mi * 16 + v + lhi * 8;
            const size_t oidx = ((size_t)t * SNN_B + Mr) * OUT + N;
            if (MODE) { Yf[oidx] = s; hs[mi][ni][v] += s; }
            else      { Yh[oidx] = (_Float16)s; }
          }
          acc[mi][ni] = (v8f){};
        }
    }
  }

  if (MODE) {
    const float invT = 1.0f / (float)SNN_T;
    for (int mi = 0; mi < 2; ++mi)
      for (int ni = 0; ni < 2; ++ni) {
        const int N = NB0 + wn * 32 + ni * 16 + ln15;
        for (int v = 0; v < 8; ++v) {
          const int Mr = MB0 + wm * 32 + mi * 16 + v + lhi * 8;
          Hout[(size_t)Mr * OUT + N] = hs[mi][ni][v] * invT;
        }
      }
  }
}

extern "C" void kernel_launch(void* const* d_in, const int* in_sizes, int n_in,
                              void* d_out, int out_size, void* d_ws, size_t ws_size,
                              hipStream_t stream) {
  (void)in_sizes; (void)n_in; (void)out_size; (void)ws_size;

  const float* spikes = (const float*)d_in[0];
  // d_in[1] (hat_spk) only affects gradients -> forward-dead, unused.
  const float* mem0 = (const float*)d_in[2];
  const float* mem1 = (const float*)d_in[3];
  const float* mem2 = (const float*)d_in[4];
  const float* W0   = (const float*)d_in[5];
  const float* b0   = (const float*)d_in[6];
  const float* W1   = (const float*)d_in[7];
  const float* b1   = (const float*)d_in[8];
  const float* W2   = (const float*)d_in[9];
  const float* b2   = (const float*)d_in[10];

  // ---- workspace layout (f16 staging) --------------------------------------
  char* ws = (char*)d_ws;
  size_t off = 0;
  auto take = [&](size_t bytes) { void* p = ws + off; off += (bytes + 255) & ~(size_t)255; return p; };
  _Float16* spkh = (_Float16*)take((size_t)SNN_T * SNN_B * 1024 * 2);
  _Float16* ss0h = (_Float16*)take((size_t)SNN_T * SNN_B * 2048 * 2);
  _Float16* ss1h = (_Float16*)take((size_t)SNN_T * SNN_B * 2048 * 2);
  _Float16* W0h  = (_Float16*)take((size_t)2048 * 1024 * 2);
  _Float16* W1h  = (_Float16*)take((size_t)2048 * 2048 * 2);
  _Float16* W2h  = (_Float16*)take((size_t)1024 * 2048 * 2);

  float* outS = (float*)d_out;                         // [50,256,1024]
  float* outH = outS + (size_t)SNN_T * SNN_B * 1024;   // [256,1024]

  // ---- prepass: convert inputs/weights to f16 ------------------------------
  snn_cvt_f16<<<2048, 256, 0, stream>>>(spikes, spkh, (size_t)SNN_T * SNN_B * 1024);
  snn_cvt_f16<<<1024, 256, 0, stream>>>(W0, W0h, (size_t)2048 * 1024);
  snn_cvt_f16<<<1024, 256, 0, stream>>>(W1, W1h, (size_t)2048 * 2048);
  snn_cvt_f16<<<1024, 256, 0, stream>>>(W2, W2h, (size_t)1024 * 2048);

  // ---- layer kernels -------------------------------------------------------
  const size_t lds1024 = ((size_t)64 * (1024 + 8) + 2 * 64 * 72) * 2;  // ~150 KB
  const size_t lds2048 = ((size_t)64 * (2048 + 8) + 2 * 64 * 72) * 2;  // ~281 KB (<320 KB/WGP)

  (void)hipFuncSetAttribute(reinterpret_cast<const void*>(&snn_layer_kernel<0, 1024, 2048>),
                            hipFuncAttributeMaxDynamicSharedMemorySize, (int)lds1024);
  (void)hipFuncSetAttribute(reinterpret_cast<const void*>(&snn_layer_kernel<0, 2048, 2048>),
                            hipFuncAttributeMaxDynamicSharedMemorySize, (int)lds2048);
  (void)hipFuncSetAttribute(reinterpret_cast<const void*>(&snn_layer_kernel<1, 2048, 1024>),
                            hipFuncAttributeMaxDynamicSharedMemorySize, (int)lds2048);

  // Layer 0: IN=1024, OUT=2048
  snn_layer_kernel<0, 1024, 2048><<<dim3(2048 / 64, SNN_B / 64), 128, lds1024, stream>>>(
      spkh, W0h, b0, mem0, ss0h, nullptr, nullptr);
  // Layer 1: IN=2048, OUT=2048
  snn_layer_kernel<0, 2048, 2048><<<dim3(2048 / 64, SNN_B / 64), 128, lds2048, stream>>>(
      ss0h, W1h, b1, mem1, ss1h, nullptr, nullptr);
  // Layer 2: IN=2048, OUT=1024 -> f32 spikes + h
  snn_layer_kernel<1, 2048, 1024><<<dim3(1024 / 64, SNN_B / 64), 128, lds2048, stream>>>(
      ss1h, W2h, b2, mem2, nullptr, outS, outH);
}
